// NonLocal_34497177322209
// MI455X (gfx1250) — matrix operand order repaired and checked by reading, not compiled, not run on hardware
//
#include <hip/hip_runtime.h>
#include <stdint.h>

// ---------------------------------------------------------------------------
// NonLocal block for MI455X (gfx1250, wave32, WMMA).
// All six GEMMs via one tiled kernel: v_wmma_f32_16x16x32_f16 (f16 in, f32
// acc). Block tile 128x128 (8 waves, 4x2; wave tile 32x64 = 2x4 WMMA accs),
// K stepped by 32 through double-buffered LDS fed by
// global_load_async_to_lds_b128 (ASYNCcnt) where the access pattern permits,
// packed-b32 LDS transpose staging where it does not. B tile held
// fragment-ready ([n][k]) so every fragment is 2x ds_load_b128 per lane.
// ---------------------------------------------------------------------------

typedef __attribute__((ext_vector_type(16))) _Float16 v16h;
typedef __attribute__((ext_vector_type(8)))  float    v8f;

#define B_  2
#define C_  1024
#define N_  6272   // T*H*W = 8*28*28
#define D_  512

#define AK_PAD 40   // lsA row stride in f16 (32 + 8): 80B rows, 16B aligned
#define BK_PAD 40   // lsB column stride in f16:       80B rows, 16B aligned

// ---------------------------- f32 -> f16 convert ---------------------------
__global__ void k_cvt_f32_f16(const float* __restrict__ in,
                              _Float16* __restrict__ out, int n) {
  int i = blockIdx.x * blockDim.x + threadIdx.x;
  if (i < n) out[i] = (_Float16)in[i];
}

// --------------------- staging helpers (one 32-deep K tile) ----------------
// Flat->LDS offset: low 32 bits of an LDS flat address are the wave-relative
// DS byte offset (ISA 00_overview aperture mapping), which is what
// global_load_async_to_lds_* VDST expects.
__device__ __forceinline__ unsigned lds_off(const void* p) {
  return (unsigned)(uintptr_t)p;
}
__device__ __forceinline__ void async_b128(unsigned lds, const _Float16* g) {
  asm volatile("global_load_async_to_lds_b128 %0, %1, off"
               :: "v"(lds), "v"((uint64_t)(uintptr_t)g) : "memory");
}
__device__ __forceinline__ void async_b128_o16(unsigned lds, const _Float16* g) {
  asm volatile("global_load_async_to_lds_b128 %0, %1, off offset:16"
               :: "v"(lds), "v"((uint64_t)(uintptr_t)g) : "memory");
}
__device__ __forceinline__ void wait_async0() {
  asm volatile("s_wait_asynccnt 0x0" ::: "memory");
}

// lsA layout: [m][k], m=0..127, stride AK_PAD (5120 f16 per tile)
template<bool TA>
__device__ __forceinline__ void stage_A(_Float16* lsA, const _Float16* A,
                                        int lda, int m0, int k0, int t) {
  if (!TA) {
    // global k-contiguous -> LDS k-contiguous: async DMA, 32B per lane
    const int row = t >> 1, kc = (t & 1) * 16;
    const _Float16* src = A + (size_t)(m0 + row) * lda + (k0 + kc);
    __builtin_prefetch((const void*)(src + 32), 0, 3);   // next K tile -> L2
    const unsigned dst = lds_off(lsA + row * AK_PAD + kc);
    async_b128(dst, src);
    async_b128_o16(dst, src);    // INST_OFFSET applies to both LDS and mem
  } else {
    // global m-contiguous: transpose via packed b32 stores (k-pairs)
    const int kp = t & 15, mg = t >> 4;      // k = 2kp,2kp+1 ; m = mg*8..+7
    const _Float16* s0 = A + (size_t)(k0 + 2 * kp) * lda + (m0 + mg * 8);
    const _Float16* s1 = s0 + lda;
    __builtin_prefetch((const void*)(s0 + (size_t)32 * lda), 0, 3);
    uint32_t* ls32 = (uint32_t*)lsA;
    #pragma unroll
    for (int i = 0; i < 8; ++i) {
      union { _Float16 h[2]; uint32_t u; } pk;
      pk.h[0] = s0[i]; pk.h[1] = s1[i];
      ls32[(mg * 8 + i) * (AK_PAD / 2) + kp] = pk.u;
    }
  }
}

// lsB layout: fragment-ready [n][k], n=0..127, stride BK_PAD
template<bool TB>
__device__ __forceinline__ void stage_B(_Float16* lsB, const _Float16* B,
                                        int ldb, int n0, int k0, int t) {
  if (TB) {
    // B[k,n] = BT[n*ldb+k]: global k-contiguous -> LDS k-contiguous: async DMA
    const int n = t >> 1, kc = (t & 1) * 16;
    const _Float16* src = B + (size_t)(n0 + n) * ldb + (k0 + kc);
    __builtin_prefetch((const void*)(src + 32), 0, 3);
    const unsigned dst = lds_off(lsB + n * BK_PAD + kc);
    async_b128(dst, src);
    async_b128_o16(dst, src);
  } else {
    // global n-contiguous: transpose via packed b32 stores (k-pairs)
    const int kp = t & 15, ng = t >> 4;      // k = 2kp,2kp+1 ; n = ng*8..+7
    const _Float16* s0 = B + (size_t)(k0 + 2 * kp) * ldb + (n0 + ng * 8);
    const _Float16* s1 = s0 + ldb;
    __builtin_prefetch((const void*)(s0 + (size_t)32 * ldb), 0, 3);
    uint32_t* ls32 = (uint32_t*)lsB;
    #pragma unroll
    for (int i = 0; i < 8; ++i) {
      union { _Float16 h[2]; uint32_t u; } pk;
      pk.h[0] = s0[i]; pk.h[1] = s1[i];
      ls32[(ng * 8 + i) * (BK_PAD / 2) + kp] = pk.u;
    }
  }
}

// ------------------------------- WMMA GEMM ---------------------------------
// Out[m,n] = scale * sum_k A[m,k]*B[k,n] (+ bias[m] if HAS_BIAS)
//   TA: A element (m,k) read as A[k*lda + m]   (else A[m*lda + k])
//   TB: B element (k,n) read as B[n*ldb + k]   (else B[k*ldb + n])
// Block: 256 threads (8 waves, 4(M) x 2(N)), tile 128(M) x 128(N),
// wave tile 32x64 (2x4 accumulators), K stepped by 32, double-buffered LDS.
// Requires M%128==0, N%128==0, K%32==0 (true for all launches below).
template<bool TA, bool TB, bool OUT16, bool HAS_BIAS>
__global__ __launch_bounds__(256) void k_gemm_wmma(
    const _Float16* __restrict__ A, int lda,
    const _Float16* __restrict__ B, int ldb,
    void* __restrict__ Out, int ldo,
    const float* __restrict__ bias, float scale,
    int M, int N, int K)
{
  __shared__ _Float16 lsA[2][128 * AK_PAD];   // 2 x 10,240 B
  __shared__ _Float16 lsB[2][128 * BK_PAD];   // 2 x 10,240 B

  constexpr bool USE_ASYNC = (!TA) || TB;

  const int t    = threadIdx.x;
  const int lane = t & 31;
  const int wave = t >> 5;
  const int wm   = wave >> 1;   // 0..3  (M direction)
  const int wn   = wave & 1;    // 0..1  (N direction)
  const int lr   = lane & 15;
  const int hi   = lane >> 4;   // lane half-group

  const int m0 = blockIdx.y * 128;
  const int n0 = blockIdx.x * 128;

  v8f acc[2][4];
  #pragma unroll
  for (int i = 0; i < 2; ++i)
    #pragma unroll
    for (int j = 0; j < 4; ++j) acc[i][j] = (v8f){};

  // prologue: stage K-tile 0 into buffer 0
  stage_A<TA>(lsA[0], A, lda, m0, 0, t);
  stage_B<TB>(lsB[0], B, ldb, n0, 0, t);
  if (USE_ASYNC) wait_async0();
  __syncthreads();

  int cur = 0;
  for (int k0 = 0; k0 < K; k0 += 32) {
    const bool has_next = (k0 + 32 < K);
    if (has_next) {                      // stage next tile into other buffer
      stage_A<TA>(lsA[cur ^ 1], A, lda, m0, k0 + 32, t);
      stage_B<TB>(lsB[cur ^ 1], B, ldb, n0, k0 + 32, t);
    }

    // ---- fragments from current buffer (ISA 05_wmma §7.12.2 layouts) ----
    v16h a[2], b[4];
    {
      const _Float16* la = lsA[cur];
      const _Float16* lb = lsB[cur];
      #pragma unroll
      for (int tm = 0; tm < 2; ++tm) {
        const int r0 = (wm * 32 + tm * 16 + lr) * AK_PAD;
        #pragma unroll
        for (int p = 0; p < 16; ++p) {
          const int v = p >> 1, e = p & 1;
          // A: contiguous K = hi*16 + p (matches B fragment order)
          const int kk = hi * 16 + p;
          a[tm][p] = la[r0 + kk];
        }
      }
      #pragma unroll
      for (int tn = 0; tn < 4; ++tn) {
        const int c0 = (wn * 64 + tn * 16 + lr) * BK_PAD + hi * 16;
        #pragma unroll
        for (int p = 0; p < 16; ++p)     // contiguous -> 2x ds_load_b128
          b[tn][p] = lb[c0 + p];
      }
    }

    #pragma unroll
    for (int tm = 0; tm < 2; ++tm)
      #pragma unroll
      for (int tn = 0; tn < 4; ++tn)
        acc[tm][tn] = __builtin_amdgcn_wmma_f32_16x16x32_f16(
            false, a[tm], false, b[tn], (short)0, acc[tm][tn], false, false);

    if (USE_ASYNC && has_next) wait_async0();   // next tile landed
    __syncthreads();                            // publish; protects reuse
    cur ^= 1;
  }

  // ---- epilogue: D layout -> VGPR r holds (M = hi*8 + r, N = lane&15) ----
  _Float16* o16 = (_Float16*)Out;
  float*    o32 = (float*)Out;
  const int mW = m0 + wm * 32;
  const int nW = n0 + wn * 64;

  float bv[2][8];
  #pragma unroll
  for (int tm = 0; tm < 2; ++tm)
    #pragma unroll
    for (int r = 0; r < 8; ++r)
      bv[tm][r] = HAS_BIAS ? bias[mW + tm * 16 + hi * 8 + r] : 0.0f;

  #pragma unroll
  for (int tm = 0; tm < 2; ++tm) {
    #pragma unroll
    for (int tn = 0; tn < 4; ++tn) {
      const int mb = mW + tm * 16 + hi * 8;
      const int nn = nW + tn * 16 + lr;
      const size_t base = (size_t)mb * ldo + nn;
      #pragma unroll
      for (int r = 0; r < 8; ++r) {
        const float val = acc[tm][tn][r] * scale + bv[tm][r];
        if (OUT16) o16[base + (size_t)r * ldo] = (_Float16)val;
        else       o32[base + (size_t)r * ldo] = val;
      }
    }
  }
}

// ------------------------- row softmax (in-place f16) ----------------------
__global__ __launch_bounds__(256) void k_softmax_rows(_Float16* __restrict__ s, int n) {
  __shared__ float red[256];
  _Float16* p = s + (size_t)blockIdx.x * n;
  const int t = threadIdx.x;

  float m = -3.0e38f;
  for (int j = t; j < n; j += 256) m = fmaxf(m, (float)p[j]);
  red[t] = m; __syncthreads();
  for (int o = 128; o > 0; o >>= 1) { if (t < o) red[t] = fmaxf(red[t], red[t + o]); __syncthreads(); }
  m = red[0]; __syncthreads();

  float sum = 0.f;
  for (int j = t; j < n; j += 256) sum += __expf((float)p[j] - m);
  red[t] = sum; __syncthreads();
  for (int o = 128; o > 0; o >>= 1) { if (t < o) red[t] += red[t + o]; __syncthreads(); }
  const float inv = 1.f / red[0];

  for (int j = t; j < n; j += 256) p[j] = (_Float16)(__expf((float)p[j] - m) * inv);
}

// ------------------ batch-norm statistics (per channel) --------------------
__global__ __launch_bounds__(256) void k_bn_stats(const float* __restrict__ z,
    const float* __restrict__ gamma, const float* __restrict__ beta,
    float* __restrict__ aOut, float* __restrict__ bOut) {
  __shared__ float rs[256], rss[256];
  const int c = blockIdx.x, t = threadIdx.x;
  float s = 0.f, ss = 0.f;
  for (int b = 0; b < B_; ++b) {
    const float* p = z + ((size_t)b * C_ + c) * N_;
    for (int j = t; j < N_; j += 256) { float v = p[j]; s += v; ss += v * v; }
  }
  rs[t] = s; rss[t] = ss; __syncthreads();
  for (int o = 128; o > 0; o >>= 1) {
    if (t < o) { rs[t] += rs[t + o]; rss[t] += rss[t + o]; }
    __syncthreads();
  }
  if (t == 0) {
    const float cnt  = (float)(B_ * N_);
    const float mean = rs[0] / cnt;
    const float var  = rss[0] / cnt - mean * mean;
    const float inv  = rsqrtf(var + 1e-5f);
    const float g    = gamma[c] * inv;
    aOut[c] = g;
    bOut[c] = beta[c] - mean * g;
  }
}

// ------------------- normalize + affine + residual -------------------------
__global__ void k_bn_apply_residual(const float* __restrict__ z, const float* __restrict__ x,
    const float* __restrict__ a, const float* __restrict__ b,
    float* __restrict__ out, size_t total) {
  size_t i = (size_t)blockIdx.x * blockDim.x + threadIdx.x;
  if (i >= total) return;
  const int c = (int)((i / N_) % C_);
  out[i] = z[i] * a[c] + b[c] + x[i];
}

// ---------------------------------------------------------------------------
extern "C" void kernel_launch(void* const* d_in, const int* in_sizes, int n_in,
                              void* d_out, int out_size, void* d_ws, size_t ws_size,
                              hipStream_t stream) {
  (void)in_sizes; (void)n_in; (void)out_size; (void)ws_size;

  const float* x       = (const float*)d_in[0];
  const float* theta_w = (const float*)d_in[1];
  const float* theta_b = (const float*)d_in[2];
  const float* phi_w   = (const float*)d_in[3];
  const float* phi_b   = (const float*)d_in[4];
  const float* g_w     = (const float*)d_in[5];
  const float* g_b     = (const float*)d_in[6];
  const float* W_w     = (const float*)d_in[7];
  const float* W_b     = (const float*)d_in[8];
  const float* gamma   = (const float*)d_in[9];
  const float* beta    = (const float*)d_in[10];

  // ---- workspace carve-out (~206 MB) ----
  char* ws = (char*)d_ws;
  size_t off = 0;
  auto alloc = [&](size_t bytes) -> char* {
    char* p = ws + off;
    off += (bytes + 255) & ~(size_t)255;
    return p;
  };
  _Float16* xh      = (_Float16*)alloc(2ULL * B_ * C_ * N_);
  _Float16* wth     = (_Float16*)alloc(2ULL * D_ * C_);
  _Float16* wph     = (_Float16*)alloc(2ULL * D_ * C_);
  _Float16* wgh     = (_Float16*)alloc(2ULL * D_ * C_);
  _Float16* wWh     = (_Float16*)alloc(2ULL * C_ * D_);
  _Float16* thetah  = (_Float16*)alloc(2ULL * B_ * D_ * N_);
  _Float16* phih    = (_Float16*)alloc(2ULL * B_ * D_ * N_);
  _Float16* ghb     = (_Float16*)alloc(2ULL * B_ * D_ * N_);
  _Float16* scoresh = (_Float16*)alloc(2ULL * N_ * N_);     // reused per batch
  _Float16* yh      = (_Float16*)alloc(2ULL * D_ * N_);     // reused per batch
  float*    zbuf    = (float*)   alloc(4ULL * B_ * C_ * N_);
  float*    abuf    = (float*)   alloc(4ULL * C_);
  float*    bbuf    = (float*)   alloc(4ULL * C_);

  // ---- precision conversion (one pass) ----
  {
    const int nx = B_ * C_ * N_;
    k_cvt_f32_f16<<<(nx + 255) / 256, 256, 0, stream>>>(x, xh, nx);
    const int nw = D_ * C_;
    k_cvt_f32_f16<<<(nw + 255) / 256, 256, 0, stream>>>(theta_w, wth, nw);
    k_cvt_f32_f16<<<(nw + 255) / 256, 256, 0, stream>>>(phi_w,   wph, nw);
    k_cvt_f32_f16<<<(nw + 255) / 256, 256, 0, stream>>>(g_w,     wgh, nw);
    k_cvt_f32_f16<<<(nw + 255) / 256, 256, 0, stream>>>(W_w,     wWh, nw);
  }

  const float inv_sqrt_d = 0.04419417382415922f;  // 512^-0.5
  const dim3 blk(256);

  for (int b = 0; b < B_; ++b) {
    const size_t xOff = (size_t)b * C_ * N_;
    const size_t pOff = (size_t)b * D_ * N_;

    // projections: [D x C] x [C x N] (+bias), f16 out
    const dim3 gp(N_ / 128, D_ / 128);
    k_gemm_wmma<false, false, true, true><<<gp, blk, 0, stream>>>(
        wth, C_, xh + xOff, N_, (void*)(thetah + pOff), N_, theta_b, 1.0f, D_, N_, C_);
    k_gemm_wmma<false, false, true, true><<<gp, blk, 0, stream>>>(
        wph, C_, xh + xOff, N_, (void*)(phih + pOff), N_, phi_b, 1.0f, D_, N_, C_);
    k_gemm_wmma<false, false, true, true><<<gp, blk, 0, stream>>>(
        wgh, C_, xh + xOff, N_, (void*)(ghb + pOff), N_, g_b, 1.0f, D_, N_, C_);

    // scores = theta^T * phi * d^-0.5  : TRANS_A, f16 out, no bias
    const dim3 gs(N_ / 128, N_ / 128);
    k_gemm_wmma<true, false, true, false><<<gs, blk, 0, stream>>>(
        thetah + pOff, N_, phih + pOff, N_, (void*)scoresh, N_,
        nullptr, inv_sqrt_d, N_, N_, D_);

    // softmax along j for each row i
    k_softmax_rows<<<N_, blk, 0, stream>>>(scoresh, N_);

    // y[d,i] = sum_j g[d,j] * attn[i,j] : A = g (D x N), B = attn^T (TRANS_B)
    const dim3 gy(N_ / 128, D_ / 128);
    k_gemm_wmma<false, true, true, false><<<gy, blk, 0, stream>>>(
        ghb + pOff, N_, scoresh, N_, (void*)yh, N_, nullptr, 1.0f, D_, N_, N_);

    // z = W * y + W_b : [C x D] x [D x N], f32 out
    const dim3 gz(N_ / 128, C_ / 128);
    k_gemm_wmma<false, false, false, true><<<gz, blk, 0, stream>>>(
        wWh, D_, yh, N_, (void*)(zbuf + xOff), N_, W_b, 1.0f, C_, N_, D_);
  }

  // batch-norm (training-mode batch stats) + affine + residual
  k_bn_stats<<<C_, blk, 0, stream>>>(zbuf, gamma, beta, abuf, bbuf);
  const size_t total = (size_t)B_ * C_ * N_;
  k_bn_apply_residual<<<(unsigned)((total + 255) / 256), blk, 0, stream>>>(
      zbuf, x, abuf, bbuf, (float*)d_out, total);
}
